// HeteroGNN_10771777979113
// MI455X (gfx1250) — matrix operand order repaired and checked by reading, not compiled, not run on hardware
//
#include <hip/hip_runtime.h>
#include <hip/hip_bf16.h>
#include <cstdint>
#include <cstddef>

// ---------------------------------------------------------------------------
// HeteroGNN forward on gfx1250 (MI455X).
// fp32 node streams in HBM (binding constraint ~5 GB @ 23.3 TB/s);
// GEMMs run on v_wmma_f32_16x16x32_bf16 with weights pre-repacked to bf16
// in WMMA B-fragment order (one contiguous 32B load per fragment).
// ---------------------------------------------------------------------------

typedef __attribute__((ext_vector_type(16))) __bf16 v16bf;
typedef __attribute__((ext_vector_type(8)))  float  v8f;

// ---- embedding concat: x_u = [age_emb | gen_emb | user_feat]  [NU x 128] ---
__global__ __launch_bounds__(256) void k_concat(
    const float* __restrict__ uf, const int* __restrict__ x1,
    const float* __restrict__ age, const float* __restrict__ gen,
    float* __restrict__ xu, int nu)
{
  int i = blockIdx.x * blockDim.x + threadIdx.x;
  int total = nu << 7;
  if (i >= total) return;
  int node = i >> 7, f = i & 127;
  float v;
  if (f < 32)      v = age[x1[2 * node + 0] * 32 + f];
  else if (f < 64) v = gen[x1[2 * node + 1] * 32 + (f - 32)];
  else             v = uf[(size_t)node * 64 + (f - 64)];
  xu[i] = v;
}

// ---- per-destination edge counts ------------------------------------------
__global__ __launch_bounds__(256) void k_count(
    const int* __restrict__ dst, float* __restrict__ cnt, int e)
{
  int t = blockIdx.x * blockDim.x + threadIdx.x;
  if (t >= e) return;
  atomicAdd(&cnt[dst[t]], 1.0f);
}

// ---- segment-sum scatter: agg[dst[e]] += X[src[e]]  (float4 per thread) ----
template <int F>
__global__ __launch_bounds__(256) void k_scatter(
    const int* __restrict__ src, const int* __restrict__ dst,
    const float* __restrict__ X, float* __restrict__ agg, int e)
{
  int t = blockIdx.x * blockDim.x + threadIdx.x;
  int total = e * (F / 4);
  if (t >= total) return;
  int ei = t / (F / 4);
  int c  = (t % (F / 4)) * 4;
  int s = src[ei], d = dst[ei];
  const float4 v = *(const float4*)(X + (size_t)s * F + c);
  float* p = agg + (size_t)d * F + c;
  atomicAdd(p + 0, v.x);
  atomicAdd(p + 1, v.y);
  atomicAdd(p + 2, v.z);
  atomicAdd(p + 3, v.w);
}

// ---- mean: agg /= max(cnt, 1) (in place) ----------------------------------
template <int F>
__global__ __launch_bounds__(256) void k_mean(
    float* __restrict__ agg, const float* __restrict__ cnt, int n)
{
  int t = blockIdx.x * blockDim.x + threadIdx.x;
  int total = n * F;
  if (t >= total) return;
  int i = t / F;
  agg[t] = agg[t] / fmaxf(cnt[i], 1.0f);
}

// ---- weight repack: f32 [K x 128] -> bf16 in WMMA B-fragment order --------
// Wp[((ks*8 + n)*32 + lane)*16 + j] = bf16( W[(ks*32 + (lane>>4)*16 + j)*128
//                                            + n*16 + (lane&15)] )
// so a lane's 16 B values for (k-step ks, n-tile n) are one contiguous 32B.
__global__ __launch_bounds__(256) void k_repack(
    const float* __restrict__ W, __bf16* __restrict__ Wp, int K)
{
  int t = blockIdx.x * blockDim.x + threadIdx.x;
  int total = K * 128;
  if (t >= total) return;
  int j    = t & 15;
  int lane = (t >> 4) & 31;
  int n    = (t >> 9) & 7;
  int ks   = t >> 12;
  int krow = ks * 32 + (lane >> 4) * 16 + j;
  int col  = n * 16 + (lane & 15);
  Wp[t] = (__bf16)W[(size_t)krow * 128 + col];
}

// ---- fused dual-input WMMA GEMM: C = relu(A1@W1 + A2@W2 + b), N = 128 -----
// One wave computes a 16x128 output tile (8 accumulators of v8f).
// A fragment (16-bit A, 16x32): lanes 0-15 -> K {0..15},
// lanes 16-31 -> K {16..31}; converted f32->bf16 in registers.
// B fragment: one contiguous v16bf load from the repacked weight buffer.
// C/D: M = r + 8*(lane>>4), N = n*16 + (lane&15).
__global__ __launch_bounds__(128) void k_gemm2_relu(
    const float* __restrict__ A1, int K1, const __bf16* __restrict__ W1,
    const float* __restrict__ A2, int K2, const __bf16* __restrict__ W2,
    const float* __restrict__ bias, float* __restrict__ C, int mtiles)
{
  const int wave = threadIdx.x >> 5;
  const int lane = threadIdx.x & 31;
  const int tile = blockIdx.x * 4 + wave;
  if (tile >= mtiles) return;          // wave-uniform: EXEC stays all-ones
  const int rlo = lane & 15;
  const int hi  = lane >> 4;
  const int row = (tile << 4) + rlo;

  v8f acc[8] = {};

  #pragma unroll
  for (int pass = 0; pass < 2; ++pass) {
    const float*  A = pass ? A2 : A1;
    const __bf16* W = pass ? W2 : W1;
    const int     K = pass ? K2 : K1;
    for (int k0 = 0; k0 < K; k0 += 32) {
      v16bf a;
      const float* ap = A + (size_t)row * K + (k0 + hi * 16);
      #pragma unroll
      for (int j = 0; j < 8; ++j) {
        a[j]     = (__bf16)ap[j];
        a[8 + j] = (__bf16)ap[8 + j];
      }
      const __bf16* wbase = W + ((size_t)(k0 >> 5) * 256 + lane) * 16;
      #pragma unroll
      for (int n = 0; n < 8; ++n) {
        v16bf b = *(const v16bf*)(wbase + (size_t)n * 512);
        acc[n] = __builtin_amdgcn_wmma_f32_16x16x32_bf16(
            false, a, false, b, (short)0, acc[n], false, false);
      }
    }
  }

  #pragma unroll
  for (int n = 0; n < 8; ++n) {
    float bv = bias[n * 16 + rlo];
    #pragma unroll
    for (int r = 0; r < 8; ++r) {
      int m = (tile << 4) + hi * 8 + r;
      float v = acc[n][r] + bv;
      C[(size_t)m * 128 + n * 16 + rlo] = v > 0.0f ? v : 0.0f;
    }
  }
}

// ---- final: sigmoid(concat(u2[mu], s2[ms]) @ lin_w + lin_b) ---------------
__global__ __launch_bounds__(256) void k_final(
    const float* __restrict__ u2, const float* __restrict__ s2,
    const int* __restrict__ mu, const int* __restrict__ ms,
    const float* __restrict__ lw, const float* __restrict__ lb,
    float* __restrict__ out, int m)
{
  int wid  = blockIdx.x * (blockDim.x >> 5) + (threadIdx.x >> 5);
  int lane = threadIdx.x & 31;
  if (wid >= m) return;                 // wave-uniform
  const float* up = u2 + (size_t)mu[wid] * 128;
  const float* sp = s2 + (size_t)ms[wid] * 128;
  float sum = 0.0f;
  #pragma unroll
  for (int j = 0; j < 4; ++j) sum += up[lane + 32 * j] * lw[lane + 32 * j];
  #pragma unroll
  for (int j = 0; j < 4; ++j) sum += sp[lane + 32 * j] * lw[128 + lane + 32 * j];
  #pragma unroll
  for (int off = 16; off > 0; off >>= 1) sum += __shfl_down(sum, off, 32);
  if (lane == 0) out[wid] = 1.0f / (1.0f + __expf(-(sum + lb[0])));
}

// ---------------------------------------------------------------------------
extern "C" void kernel_launch(void* const* d_in, const int* in_sizes, int n_in,
                              void* d_out, int out_size, void* d_ws, size_t ws_size,
                              hipStream_t stream)
{
  const float* x_user   = (const float*)d_in[0];
  const float* x_seller = (const float*)d_in[1];
  const int*   x1       = (const int*)  d_in[2];
  const int*   src_user = (const int*)  d_in[3];
  const int*   dst_sell = (const int*)  d_in[4];
  const int*   src_sell = (const int*)  d_in[5];
  const int*   dst_user = (const int*)  d_in[6];
  const int*   mask_u   = (const int*)  d_in[7];
  const int*   mask_s   = (const int*)  d_in[8];
  const float* age_emb  = (const float*)d_in[9];
  const float* gen_emb  = (const float*)d_in[10];
  const float* W1ul = (const float*)d_in[11];
  const float* W1ur = (const float*)d_in[12];
  const float* b1u  = (const float*)d_in[13];
  const float* W1sl = (const float*)d_in[14];
  const float* W1sr = (const float*)d_in[15];
  const float* b1s  = (const float*)d_in[16];
  const float* W2ul = (const float*)d_in[17];
  const float* W2ur = (const float*)d_in[18];
  const float* b2u  = (const float*)d_in[19];
  const float* W2sl = (const float*)d_in[20];
  const float* W2sr = (const float*)d_in[21];
  const float* b2s  = (const float*)d_in[22];
  const float* lin_w = (const float*)d_in[23];
  const float* lin_b = (const float*)d_in[24];
  float* out = (float*)d_out;

  const int NU = in_sizes[0] / 64;
  const int NS = in_sizes[1] / 64;
  const int E  = in_sizes[3];
  const int M  = in_sizes[7];

  // -------- workspace layout (bytes) ---------------------------------------
  char* w = (char*)d_ws;
  const size_t szXU   = (size_t)NU * 128 * 4;   // 102.4 MB
  const size_t szAGGU = (size_t)NU * 64  * 4;   //  51.2 MB
  const size_t szAGGS = (size_t)NS * 128 * 4;   //  25.6 MB
  float* xu    = (float*)(w);                                   // x_u, later agg2_u/mean2_u
  float* aggu  = (float*)(w + szXU);                            // agg_u/mean_u; later mean2_s + s2
  float* aggs  = (float*)(w + szXU + szAGGU);                   // agg_s/mean_s
  float* u1    = (float*)(w + szXU + szAGGU + szAGGS);
  float* s1    = (float*)(w + szXU + szAGGU + szAGGS + szXU);
  float* u2    = (float*)(w + szXU + szAGGU + szAGGS + szXU + szAGGS);
  float* cntu  = (float*)(w + 2 * szXU + szAGGU + 2 * szAGGS + szXU);
  float* cnts  = cntu + NU;
  float* mean2s = aggu;                 // NS*128 floats (fits: NU*64 == 2*NS*128)
  float* s2     = aggu + (size_t)NS * 128;

  // repacked bf16 weights (fragment order), after the counters, 256B aligned
  size_t offWp = 2 * szXU + szAGGU + 2 * szAGGS + szXU + (size_t)(NU + NS) * 4;
  offWp = (offWp + 255) & ~(size_t)255;
  __bf16* wp1ul = (__bf16*)(w + offWp);               //  64*128
  __bf16* wp1ur = wp1ul +  64 * 128;                  // 128*128
  __bf16* wp1sl = wp1ur + 128 * 128;                  // 128*128
  __bf16* wp1sr = wp1sl + 128 * 128;                  //  64*128
  __bf16* wp2ul = wp1sr +  64 * 128;                  // 128*128
  __bf16* wp2ur = wp2ul + 128 * 128;
  __bf16* wp2sl = wp2ur + 128 * 128;
  __bf16* wp2sr = wp2sl + 128 * 128;

  const int mtU = NU / 16;              // 12500 (exact)
  const int mtS = NS / 16;              // 3125  (exact)

  // -------- stage 0: weight repack + embedding concat ----------------------
  k_repack<<<( 64 * 128 + 255) / 256, 256, 0, stream>>>(W1ul, wp1ul,  64);
  k_repack<<<(128 * 128 + 255) / 256, 256, 0, stream>>>(W1ur, wp1ur, 128);
  k_repack<<<(128 * 128 + 255) / 256, 256, 0, stream>>>(W1sl, wp1sl, 128);
  k_repack<<<( 64 * 128 + 255) / 256, 256, 0, stream>>>(W1sr, wp1sr,  64);
  k_repack<<<(128 * 128 + 255) / 256, 256, 0, stream>>>(W2ul, wp2ul, 128);
  k_repack<<<(128 * 128 + 255) / 256, 256, 0, stream>>>(W2ur, wp2ur, 128);
  k_repack<<<(128 * 128 + 255) / 256, 256, 0, stream>>>(W2sl, wp2sl, 128);
  k_repack<<<(128 * 128 + 255) / 256, 256, 0, stream>>>(W2sr, wp2sr, 128);
  {
    int total = NU * 128;
    k_concat<<<(total + 255) / 256, 256, 0, stream>>>(x_user, x1, age_emb, gen_emb, xu, NU);
  }

  // -------- stage 1: counts + layer-1 aggregation --------------------------
  hipMemsetAsync(cntu, 0, (size_t)NU * 4, stream);
  hipMemsetAsync(cnts, 0, (size_t)NS * 4, stream);
  hipMemsetAsync(aggu, 0, szAGGU, stream);
  hipMemsetAsync(aggs, 0, szAGGS, stream);
  k_count<<<(E + 255) / 256, 256, 0, stream>>>(dst_user, cntu, E);
  k_count<<<(E + 255) / 256, 256, 0, stream>>>(dst_sell, cnts, E);
  {
    int t64  = E * 16;   // E * (64/4)
    int t128 = E * 32;   // E * (128/4)
    k_scatter<64> <<<(t64  + 255) / 256, 256, 0, stream>>>(src_sell, dst_user, x_seller, aggu, E);
    k_scatter<128><<<(t128 + 255) / 256, 256, 0, stream>>>(src_user, dst_sell, xu,       aggs, E);
    k_mean<64> <<<((NU * 64)  + 255) / 256, 256, 0, stream>>>(aggu, cntu, NU);
    k_mean<128><<<((NS * 128) + 255) / 256, 256, 0, stream>>>(aggs, cnts, NS);
  }

  // -------- stage 2: layer-1 GEMMs (WMMA) ----------------------------------
  k_gemm2_relu<<<(mtU + 3) / 4, 128, 0, stream>>>(aggu, 64,  wp1ul, xu,       128, wp1ur, b1u, u1, mtU);
  k_gemm2_relu<<<(mtS + 3) / 4, 128, 0, stream>>>(aggs, 128, wp1sl, x_seller, 64,  wp1sr, b1s, s1, mtS);

  // -------- stage 3: layer-2 aggregation (reuse xu and aggu slots) ---------
  hipMemsetAsync(xu,     0, szXU,   stream);   // agg2_u  [NU x 128]
  hipMemsetAsync(mean2s, 0, szAGGS, stream);   // agg2_s  [NS x 128]
  {
    int t128 = E * 32;
    k_scatter<128><<<(t128 + 255) / 256, 256, 0, stream>>>(src_sell, dst_user, s1, xu,     E);
    k_scatter<128><<<(t128 + 255) / 256, 256, 0, stream>>>(src_user, dst_sell, u1, mean2s, E);
    k_mean<128><<<((NU * 128) + 255) / 256, 256, 0, stream>>>(xu,     cntu, NU);
    k_mean<128><<<((NS * 128) + 255) / 256, 256, 0, stream>>>(mean2s, cnts, NS);
  }

  // -------- stage 4: layer-2 GEMMs (WMMA) ----------------------------------
  k_gemm2_relu<<<(mtU + 3) / 4, 128, 0, stream>>>(xu,     128, wp2ul, u1, 128, wp2ur, b2u, u2, mtU);
  k_gemm2_relu<<<(mtS + 3) / 4, 128, 0, stream>>>(mean2s, 128, wp2sl, s1, 128, wp2sr, b2s, s2, mtS);

  // -------- stage 5: masked gather + linear + sigmoid ----------------------
  {
    int wavesPerBlock = 8;  // 256 threads
    int blocks = (M + wavesPerBlock - 1) / wavesPerBlock;
    k_final<<<blocks, 256, 0, stream>>>(u2, s2, mask_u, mask_s, lin_w, lin_b, out, M);
  }
}
